// HeteroGCN_69930657513814
// MI455X (gfx1250) — compile-verified, matrix operand order checked by reading, not run on hardware
//
#include <hip/hip_runtime.h>
#include <hip/hip_bf16.h>

// ---------------------------------------------------------------------------
// HeteroGCN (2-layer hetero SAGE) for gfx1250 / MI455X.
//  - scatter-mean aggregation with fp32 global atomics
//  - weights pre-converted once to bf16 in transposed (N x K) fragment layout
//  - dense projections with v_wmma_f32_16x16x32_bf16 (fp32 accumulate),
//    B fragments loaded straight from global (L2-resident), A tile staged in
//    double-buffered LDS with one barrier per k-step
//  - bias + ReLU fused epilogue
// ---------------------------------------------------------------------------

typedef __attribute__((ext_vector_type(16))) __bf16 v16bf;
typedef __attribute__((ext_vector_type(8)))  float  v8f;

union Frag16 { uint4 u[2]; v16bf v; };

__device__ __forceinline__ unsigned short f2bf(float f) {
  union { float f; unsigned int u; } c;
  c.f = f;
  unsigned int r = c.u + 0x7FFFu + ((c.u >> 16) & 1u);  // round-to-nearest-even
  return (unsigned short)(r >> 16);
}

#define ND 4000
#define NDIS 8000
#define NP 20000
#define FIN 256
#define HID 256
#define OUTF 128

// ---------------------------------------------------------------------------
// Edge count: cnt[dst] += 1  (one thread per edge)
// ---------------------------------------------------------------------------
__global__ void count_kernel(const int* __restrict__ ei, int E, float* __restrict__ cnt) {
  int e = blockIdx.x * 256 + threadIdx.x;
  if (e < E) atomicAdd(&cnt[ei[E + e]], 1.0f);
}

// ---------------------------------------------------------------------------
// Scatter-add features: agg[dst, :] += x[src, :]   (one block of 256 = F per edge)
// ---------------------------------------------------------------------------
__global__ void scatter_kernel(const float* __restrict__ x, const int* __restrict__ ei,
                               int E, float* __restrict__ agg) {
  int e = blockIdx.x;
  int t = threadIdx.x;
  int src = ei[e];
  int dst = ei[E + e];
  atomicAdd(&agg[(size_t)dst * 256 + t], x[(size_t)src * 256 + t]);
}

// ---------------------------------------------------------------------------
// One-time W conversion: fp32 W[r][k][n] (K=256 x N) -> bf16 Wt[r][n][k]
// Tiled 16x16 transpose, coalesced both directions. grid = (16, N/16, R)
// ---------------------------------------------------------------------------
__global__ void wtrans_kernel(const float* __restrict__ W, unsigned short* __restrict__ Wt,
                              int N) {
  __shared__ unsigned short tile[16][17];
  const int r = blockIdx.z;
  const float* Ws = W + (size_t)r * 256 * N;
  unsigned short* Wts = Wt + (size_t)r * (size_t)N * 256;
  const int k0 = blockIdx.x * 16, n0 = blockIdx.y * 16;
  const int t = threadIdx.x;
  tile[t >> 4][t & 15] = f2bf(Ws[(size_t)(k0 + (t >> 4)) * N + (n0 + (t & 15))]);
  __syncthreads();
  Wts[(size_t)(n0 + (t >> 4)) * 256 + k0 + (t & 15)] = tile[t & 15][t >> 4];
}

// ---------------------------------------------------------------------------
// WMMA GEMM:  out[M x N] = relu?( sum_s (A_s * rcp(cnt_s)?) @ W_s + sum_j bias_j )
// K fixed at 256. Block = 256 threads (8 waves), computes 16 rows x N cols.
// Wt is bf16, transposed [N][256] per source (B-fragment friendly).
// ---------------------------------------------------------------------------
struct GemmArgs {
  const float*          A[6];     // M x 256, fp32
  const float*          Acnt[6];  // per-row divisor (mean), nullptr => no divide
  const unsigned short* Wt[6];    // N x 256, bf16, transposed
  const float*          bias[3];  // N
  float*                out;      // M x N
  int relu;
};

template <int N, int NSRC, int NBIAS>
__global__ __launch_bounds__(256, 2) void gemm_wmma_kernel(GemmArgs g) {
  constexpr int TPW = (N >> 4) >> 3;   // column tiles per wave: 2 (N=256) / 1 (N=128)

  __shared__ __align__(16) unsigned short aS[2][16 * 32];  // double-buffered A tile (bf16)

  const int tid  = threadIdx.x;
  const int lane = tid & 31;
  const int wave = tid >> 5;
  const int lh   = lane >> 4;   // lane half (0: lanes 0-15, 1: lanes 16-31)
  const int l15  = lane & 15;
  const int rowbase = blockIdx.x * 16;

  // A-staging coordinates: thread tid stages row aR, cols aC..aC+1 of each k-tile
  const int aR = tid >> 4;
  const int aC = (tid & 15) * 2;

  v8f acc[TPW];
  #pragma unroll
  for (int t = 0; t < TPW; ++t)
    acc[t] = (v8f){0.f, 0.f, 0.f, 0.f, 0.f, 0.f, 0.f, 0.f};

  int kk = 0;  // global k-step counter across sources (LDS buffer parity)

  #pragma unroll
  for (int s = 0; s < NSRC; ++s) {
    const float* __restrict__ Ap           = g.A[s];
    const float* __restrict__ Cp           = g.Acnt[s];
    const unsigned short* __restrict__ Wtp = g.Wt[s];

    // mean-divide scale for this thread's staging row (hoisted out of k loop)
    float scale = 1.0f;
    if (Cp) scale = 1.0f / fmaxf(Cp[rowbase + aR], 1.0f);

    #pragma unroll
    for (int k = 0; k < 256; k += 32, ++kk) {
      unsigned short* aBuf = aS[kk & 1];

      // ---- stage A tile (16 rows x 32 k): one float2 per thread ----
      {
        const float2 av = *(const float2*)&Ap[(size_t)(rowbase + aR) * 256 + (k + aC)];
        aBuf[aR * 32 + aC]     = f2bf(av.x * scale);
        aBuf[aR * 32 + aC + 1] = f2bf(av.y * scale);
        if (k + 32 < 256)  // next A k-tile -> global_prefetch_b8
          __builtin_prefetch(&Ap[(size_t)(rowbase + aR) * 256 + k + 32 + aC], 0, 1);
      }
      __syncthreads();

      // ---- A fragment from LDS: lane = row M, two contiguous 8-half chunks ----
      Frag16 af;
      const uint4* ap4 = (const uint4*)aBuf;        // 4 uint4 per 32-half row
      af.u[0] = ap4[l15 * 4 + lh];                  // halfs 0-7 : k = 8*lh + 0..7
      af.u[1] = ap4[l15 * 4 + 2 + lh];              // halfs 8-15: k = 16 + 8*lh + 0..7

      // ---- B fragments straight from global bf16 Wt (L2-resident) ----
      #pragma unroll
      for (int t = 0; t < TPW; ++t) {
        const int n = (wave + (t << 3)) * 16 + l15;  // output column for this lane
        Frag16 bfr;
        bfr.u[0] = *(const uint4*)&Wtp[(size_t)n * 256 + k + 8 * lh];
        bfr.u[1] = *(const uint4*)&Wtp[(size_t)n * 256 + k + 16 + 8 * lh];
        acc[t] = __builtin_amdgcn_wmma_f32_16x16x32_bf16(
            false, af.v, false, bfr.v, (short)0, acc[t], false, false);
      }
    }
  }

  // ---- epilogue: C layout (VGPR v: M = v + 8*lh, N = lane&15) ----
  #pragma unroll
  for (int t = 0; t < TPW; ++t) {
    const int col = (wave + (t << 3)) * 16 + l15;
    float bsum = 0.f;
    #pragma unroll
    for (int j = 0; j < NBIAS; ++j) bsum += g.bias[j][col];
    #pragma unroll
    for (int v = 0; v < 8; ++v) {
      float o = acc[t][v] + bsum;
      if (g.relu) o = fmaxf(o, 0.f);
      g.out[(size_t)(rowbase + lh * 8 + v) * N + col] = o;
    }
  }
}

// ---------------------------------------------------------------------------
extern "C" void kernel_launch(void* const* d_in, const int* in_sizes, int n_in,
                              void* d_out, int out_size, void* d_ws, size_t ws_size,
                              hipStream_t stream) {
  const float* x_d   = (const float*)d_in[0];
  const float* x_dis = (const float*)d_in[1];
  const float* x_p   = (const float*)d_in[2];
  const int* ei_dp   = (const int*)d_in[3];
  const int* ei_pd   = (const int*)d_in[4];
  const int* ei_pdis = (const int*)d_in[5];
  const int* ei_disp = (const int*)d_in[6];
  const int* ei_pp   = (const int*)d_in[7];
  const float* Wl1 = (const float*)d_in[8];
  const float* Wr1 = (const float*)d_in[9];
  const float* b1  = (const float*)d_in[10];
  const float* Wl2 = (const float*)d_in[11];
  const float* Wr2 = (const float*)d_in[12];
  const float* b2  = (const float*)d_in[13];

  const int E_dp   = in_sizes[3] / 2;
  const int E_pd   = in_sizes[4] / 2;
  const int E_pdis = in_sizes[5] / 2;
  const int E_disp = in_sizes[6] / 2;
  const int E_pp   = in_sizes[7] / 2;

  // ---- workspace layout ----
  float* ws = (float*)d_ws;
  const size_t AGG_ROWS = 3 * (size_t)NP + ND + NDIS;  // 72000
  float* agg_p_dp   = ws;
  float* agg_p_disp = agg_p_dp   + (size_t)NP * 256;
  float* agg_p_pp   = agg_p_disp + (size_t)NP * 256;
  float* agg_d      = agg_p_pp   + (size_t)NP * 256;
  float* agg_dis    = agg_d      + (size_t)ND * 256;
  float* cnt        = ws + AGG_ROWS * 256;
  float* cnt_p_dp   = cnt;
  float* cnt_p_disp = cnt + NP;
  float* cnt_p_pp   = cnt + 2 * NP;
  float* cnt_d      = cnt + 3 * NP;
  float* cnt_dis    = cnt + 3 * NP + ND;
  float* hd   = cnt + AGG_ROWS;
  float* hdis = hd   + (size_t)ND * 256;
  float* hp   = hdis + (size_t)NDIS * 256;
  // bf16 transposed weights (after fp32 region; 16B-aligned by construction)
  unsigned short* wl1t = (unsigned short*)(hp + (size_t)NP * 256);
  unsigned short* wr1t = wl1t + 5 * (size_t)FIN * HID;
  unsigned short* wl2t = wr1t + 5 * (size_t)FIN * HID;
  unsigned short* wr2t = wl2t + 5 * (size_t)HID * OUTF;

  float* od   = (float*)d_out;
  float* odis = od   + (size_t)ND * OUTF;
  float* op   = odis + (size_t)NDIS * OUTF;

  const size_t KN1 = (size_t)FIN * HID;   // 65536
  const size_t KN2 = (size_t)HID * OUTF;  // 32768

  // ---- one-time weight convert + transpose (bf16, [N][K] layout) ----
  wtrans_kernel<<<dim3(16, HID / 16, 5), 256, 0, stream>>>(Wl1, wl1t, HID);
  wtrans_kernel<<<dim3(16, HID / 16, 5), 256, 0, stream>>>(Wr1, wr1t, HID);
  wtrans_kernel<<<dim3(16, OUTF / 16, 5), 256, 0, stream>>>(Wl2, wl2t, OUTF);
  wtrans_kernel<<<dim3(16, OUTF / 16, 5), 256, 0, stream>>>(Wr2, wr2t, OUTF);

  // ---- counts (edge-structure only; shared by both layers) ----
  hipMemsetAsync(cnt, 0, AGG_ROWS * sizeof(float), stream);
  count_kernel<<<(E_dp + 255) / 256, 256, 0, stream>>>(ei_dp, E_dp, cnt_p_dp);
  count_kernel<<<(E_disp + 255) / 256, 256, 0, stream>>>(ei_disp, E_disp, cnt_p_disp);
  count_kernel<<<(E_pp + 255) / 256, 256, 0, stream>>>(ei_pp, E_pp, cnt_p_pp);
  count_kernel<<<(E_pd + 255) / 256, 256, 0, stream>>>(ei_pd, E_pd, cnt_d);
  count_kernel<<<(E_pdis + 255) / 256, 256, 0, stream>>>(ei_pdis, E_pdis, cnt_dis);

  // ================= layer 1 =================
  hipMemsetAsync(ws, 0, AGG_ROWS * 256 * sizeof(float), stream);
  scatter_kernel<<<E_dp, 256, 0, stream>>>(x_d, ei_dp, E_dp, agg_p_dp);
  scatter_kernel<<<E_disp, 256, 0, stream>>>(x_dis, ei_disp, E_disp, agg_p_disp);
  scatter_kernel<<<E_pp, 256, 0, stream>>>(x_p, ei_pp, E_pp, agg_p_pp);
  scatter_kernel<<<E_pd, 256, 0, stream>>>(x_p, ei_pd, E_pd, agg_d);
  scatter_kernel<<<E_pdis, 256, 0, stream>>>(x_p, ei_pdis, E_pdis, agg_dis);

  {  // drug: rel 1 (pd)
    GemmArgs g{};
    g.A[0] = agg_d; g.Acnt[0] = cnt_d;   g.Wt[0] = wl1t + 1 * KN1;
    g.A[1] = x_d;   g.Acnt[1] = nullptr; g.Wt[1] = wr1t + 1 * KN1;
    g.bias[0] = b1 + 1 * HID;
    g.relu = 1; g.out = hd;
    gemm_wmma_kernel<256, 2, 1><<<ND / 16, 256, 0, stream>>>(g);
  }
  {  // disease: rel 2 (pdis)
    GemmArgs g{};
    g.A[0] = agg_dis; g.Acnt[0] = cnt_dis; g.Wt[0] = wl1t + 2 * KN1;
    g.A[1] = x_dis;   g.Acnt[1] = nullptr; g.Wt[1] = wr1t + 2 * KN1;
    g.bias[0] = b1 + 2 * HID;
    g.relu = 1; g.out = hdis;
    gemm_wmma_kernel<256, 2, 1><<<NDIS / 16, 256, 0, stream>>>(g);
  }
  {  // protein: rels 0 (dp), 3 (disp), 4 (pp)
    GemmArgs g{};
    g.A[0] = agg_p_dp;   g.Acnt[0] = cnt_p_dp;   g.Wt[0] = wl1t + 0 * KN1;
    g.A[1] = agg_p_disp; g.Acnt[1] = cnt_p_disp; g.Wt[1] = wl1t + 3 * KN1;
    g.A[2] = agg_p_pp;   g.Acnt[2] = cnt_p_pp;   g.Wt[2] = wl1t + 4 * KN1;
    g.A[3] = x_p; g.Acnt[3] = nullptr; g.Wt[3] = wr1t + 0 * KN1;
    g.A[4] = x_p; g.Acnt[4] = nullptr; g.Wt[4] = wr1t + 3 * KN1;
    g.A[5] = x_p; g.Acnt[5] = nullptr; g.Wt[5] = wr1t + 4 * KN1;
    g.bias[0] = b1 + 0 * HID; g.bias[1] = b1 + 3 * HID; g.bias[2] = b1 + 4 * HID;
    g.relu = 1; g.out = hp;
    gemm_wmma_kernel<256, 6, 3><<<NP / 16, 256, 0, stream>>>(g);
  }

  // ================= layer 2 =================
  hipMemsetAsync(ws, 0, AGG_ROWS * 256 * sizeof(float), stream);
  scatter_kernel<<<E_dp, 256, 0, stream>>>(hd, ei_dp, E_dp, agg_p_dp);
  scatter_kernel<<<E_disp, 256, 0, stream>>>(hdis, ei_disp, E_disp, agg_p_disp);
  scatter_kernel<<<E_pp, 256, 0, stream>>>(hp, ei_pp, E_pp, agg_p_pp);
  scatter_kernel<<<E_pd, 256, 0, stream>>>(hp, ei_pd, E_pd, agg_d);
  scatter_kernel<<<E_pdis, 256, 0, stream>>>(hp, ei_pdis, E_pdis, agg_dis);

  {  // drug out
    GemmArgs g{};
    g.A[0] = agg_d; g.Acnt[0] = cnt_d;   g.Wt[0] = wl2t + 1 * KN2;
    g.A[1] = hd;    g.Acnt[1] = nullptr; g.Wt[1] = wr2t + 1 * KN2;
    g.bias[0] = b2 + 1 * OUTF;
    g.relu = 0; g.out = od;
    gemm_wmma_kernel<128, 2, 1><<<ND / 16, 256, 0, stream>>>(g);
  }
  {  // disease out
    GemmArgs g{};
    g.A[0] = agg_dis; g.Acnt[0] = cnt_dis; g.Wt[0] = wl2t + 2 * KN2;
    g.A[1] = hdis;    g.Acnt[1] = nullptr; g.Wt[1] = wr2t + 2 * KN2;
    g.bias[0] = b2 + 2 * OUTF;
    g.relu = 0; g.out = odis;
    gemm_wmma_kernel<128, 2, 1><<<NDIS / 16, 256, 0, stream>>>(g);
  }
  {  // protein out
    GemmArgs g{};
    g.A[0] = agg_p_dp;   g.Acnt[0] = cnt_p_dp;   g.Wt[0] = wl2t + 0 * KN2;
    g.A[1] = agg_p_disp; g.Acnt[1] = cnt_p_disp; g.Wt[1] = wl2t + 3 * KN2;
    g.A[2] = agg_p_pp;   g.Acnt[2] = cnt_p_pp;   g.Wt[2] = wl2t + 4 * KN2;
    g.A[3] = hp; g.Acnt[3] = nullptr; g.Wt[3] = wr2t + 0 * KN2;
    g.A[4] = hp; g.Acnt[4] = nullptr; g.Wt[4] = wr2t + 3 * KN2;
    g.A[5] = hp; g.Acnt[5] = nullptr; g.Wt[5] = wr2t + 4 * KN2;
    g.bias[0] = b2 + 0 * OUTF; g.bias[1] = b2 + 3 * OUTF; g.bias[2] = b2 + 4 * OUTF;
    g.relu = 0; g.out = op;
    gemm_wmma_kernel<128, 6, 3><<<NP / 16, 256, 0, stream>>>(g);
  }
}
